// EdgeFocusedGraphNetwork_69801808495210
// MI455X (gfx1250) — matrix operand order repaired and checked by reading, not compiled, not run on hardware
//
#include <hip/hip_runtime.h>

#define B_     8
#define L_     128
#define H_     256
#define NINP_  512
#define ROWS_V (B_ * L_)        // 1024
#define ROWS_E (B_ * L_ * L_)   // 131072

typedef __attribute__((ext_vector_type(16))) __bf16 v16bf;
typedef __attribute__((ext_vector_type(8)))  float  v8f;

#define GM_PLAIN 0
#define GM_EDGE  1
#define GM_ACC   2

// ---------------------------------------------------------------------------
// gfx1250 async global->LDS DMA (ASYNCcnt-tracked), ISA cdna5 §08.
//   lds_off : LDS byte address (low 32 bits of generic LDS pointer)
//   gbase   : uniform 64-bit global base (SADDR), goff: per-lane byte offset
// ---------------------------------------------------------------------------
static __device__ __forceinline__ void async_copy16(unsigned lds_off,
                                                    const void* gbase,
                                                    unsigned goff) {
    asm volatile("global_load_async_to_lds_b128 %0, %1, %2"
                 :
                 : "v"(lds_off), "v"(goff),
                   "s"((unsigned long long)(size_t)gbase)
                 : "memory");
}
static __device__ __forceinline__ void async_wait0() {
    asm volatile("s_wait_asynccnt 0" ::: "memory");
}
static __device__ __forceinline__ unsigned lds_addr(const void* p) {
    return (unsigned)(size_t)p;     // LDS aperture: addr[31:0] == LDS offset
}

// ---------------------------------------------------------------------------
// Generic WMMA GEMM on bf16 inputs:  C[M,N] = epilogue( A[M,K] @ W[N,K]^T )
//   A,W bf16 row-major (strides lda/ldw, elements); C f32 or bf16 (out_bf16).
//   GM_PLAIN: C = rowscale[m]*(A@W^T) + bias[n]
//   GM_EDGE : C = A@W^T + px[b,i][n] + py[b,j][n] + cvec[n]    (fe update)
//   GM_ACC  : C += A@W^T
// 256 thr = 8 wave32; 64x64 tile; K chunks of 32 staged by async 128b DMA.
// ---------------------------------------------------------------------------
__launch_bounds__(256)
__global__ void wmma_gemm_kernel(const __bf16* __restrict__ A, int lda,
                                 const __bf16* __restrict__ W, int ldw,
                                 const float* __restrict__ bias,
                                 const float* __restrict__ rowscale,
                                 void* __restrict__ Cv, int ldc, int out_bf16,
                                 int K, int mode,
                                 const __bf16* __restrict__ px,
                                 const __bf16* __restrict__ py,
                                 const float* __restrict__ cvec)
{
    __shared__ __bf16 As[64][40];   // 64 rows x 32 k (+8 pad) -> 80B row, 16B aligned
    __shared__ __bf16 Ws[64][40];

    const int tid  = threadIdx.x;
    const int lane = tid & 31;
    const int wv   = tid >> 5;                 // 0..7
    const int row0 = blockIdx.y * 64;
    const int col0 = blockIdx.x * 64;
    const int mrow  = (wv & 3) * 16;
    const int nbase = (wv >> 2) * 32;

    const unsigned asB = lds_addr(&As[0][0]);
    const unsigned wsB = lds_addr(&Ws[0][0]);
    const int r16 = tid >> 2;                  // tile row 0..63
    const int c16 = (tid & 3) * 16;            // 16B chunk within 64B row

    v8f acc[2] = {};

    const int nchunk = K >> 5;
    for (int kc = 0; kc < nchunk; ++kc) {
        const int k0 = kc << 5;
        // one 128-bit async DMA per thread per matrix: 64 rows x 64 bytes
        async_copy16(asB + (unsigned)(r16 * 80 + c16), A,
                     (unsigned)((size_t)(row0 + r16) * lda * 2 + k0 * 2 + c16));
        async_copy16(wsB + (unsigned)(r16 * 80 + c16), W,
                     (unsigned)((size_t)(col0 + r16) * ldw * 2 + k0 * 2 + c16));
        if (kc + 1 < nchunk)
            __builtin_prefetch(&A[(size_t)(row0 + r16) * lda + k0 + 32], 0, 1);
        async_wait0();
        __syncthreads();

        // A fragment, CDNA5 16-bit 16x32 layout (two contiguous 16B runs)
        v16bf af;
        {
            int m  = mrow + (lane & 15);
            int kb = (lane >> 4) << 3;
            #pragma unroll
            for (int e = 0; e < 16; ++e) {
                int k = (e < 8) ? (kb + e) : (kb + e + 8);
                af[e] = As[m][k];
            }
        }
        #pragma unroll
        for (int t = 0; t < 2; ++t) {
            v16bf bfrg;
            int n  = nbase + t * 16 + (lane & 15);
            int kb = (lane >> 4) << 3;
            #pragma unroll
            for (int e = 0; e < 16; ++e) bfrg[e] = Ws[n][(e < 8) ? (kb + e) : (kb + e + 8)];
            acc[t] = __builtin_amdgcn_wmma_f32_16x16x32_bf16(
                false, af, false, bfrg, (short)0, acc[t], false, false);
        }
        __syncthreads();
    }

    // epilogue, CDNA5 f32 C layout: VGPR r -> M = r + 8*(lane>=16), N = lane&15
    #pragma unroll
    for (int t = 0; t < 2; ++t) {
        #pragma unroll
        for (int r = 0; r < 8; ++r) {
            int m = row0 + mrow + r + ((lane >> 4) << 3);
            int n = col0 + nbase + t * 16 + (lane & 15);
            float v = acc[t][r];
            if (rowscale) v *= rowscale[m];
            if (bias)     v += bias[n];
            if (mode == GM_EDGE)
                v += (float)px[(m >> 7) * H_ + n]
                   + (float)py[(((m >> 14) << 7) + (m & 127)) * H_ + n]
                   + cvec[n];
            size_t ci = (size_t)m * ldc + n;
            if (out_bf16) {
                __bf16* Cb = (__bf16*)Cv;
                if (mode == GM_ACC) v += (float)Cb[ci];
                Cb[ci] = (__bf16)v;
            } else {
                float* Cf = (float*)Cv;
                if (mode == GM_ACC) v += Cf[ci];
                Cf[ci] = v;
            }
        }
    }
}

// fe[b,i,j,h] = xx[b,i,h] + yy[b,j,h] + b_fe[h]   (bf16 out)
__global__ void outer_sum_kernel(const __bf16* __restrict__ xx,
                                 const __bf16* __restrict__ yy,
                                 const float* __restrict__ bfe,
                                 __bf16* __restrict__ fe)
{
    unsigned g   = blockIdx.x * 256u + threadIdx.x;
    int      h   = g & (H_ - 1);
    unsigned row = g >> 8;
    unsigned xi  = row >> 7;                              // b*L + i
    unsigned yj  = ((row >> 14) << 7) | (row & 127);      // b*L + j
    fe[g] = (__bf16)((float)xx[xi * H_ + h] + (float)yy[yj * H_ + h] + bfe[h]);
}

// cvec[n] = b_fe @ W_ue[:, :H]^T + b_ue   (tiny, f32, one block)
__global__ void cvec_kernel(const float* __restrict__ bfe,
                            const float* __restrict__ Wue,
                            const float* __restrict__ bue,
                            float* __restrict__ cvec)
{
    int n = threadIdx.x;
    float s = bue[n];
    for (int k = 0; k < H_; ++k) s += bfe[k] * Wue[n * (2 * H_) + k];
    cvec[n] = s;
}

// f32 -> bf16 conversion (weights / feat, once per launch)
__global__ void cvt_bf16_kernel(const float* __restrict__ src,
                                __bf16* __restrict__ dst, int n)
{
    int g = blockIdx.x * 256 + threadIdx.x;
    if (g < n) dst[g] = (__bf16)src[g];
}

// ---------------------------------------------------------------------------
// Fused: attn = fe[b,:,j,:] @ W_attn^T + b_attn (WMMA over LDS slab),
//        softmax over i, agg = sigmoid(sum_i w * xhid). 216 KB dynamic LDS.
// ---------------------------------------------------------------------------
#define FES_STRIDE 264            // bf16 elems, 528B rows (16B multiple)
#define ATT_STRIDE 260            // f32 elems
#define WL_STRIDE  40             // bf16 elems, 80B rows
#define ATTN_LDS_BYTES (L_ * FES_STRIDE * 2 + L_ * ATT_STRIDE * 4 + H_ * WL_STRIDE * 2)

__launch_bounds__(256)
__global__ void attn_softmax_agg_kernel(const __bf16* __restrict__ fe,
                                        const __bf16* __restrict__ Wattn,
                                        const float* __restrict__ battn,
                                        const __bf16* __restrict__ xhid,
                                        __bf16* __restrict__ agg)
{
    extern __shared__ unsigned char smem[];
    __bf16* feS  = (__bf16*)smem;                                        // [128][264]
    float*  attS = (float*)(smem + L_ * FES_STRIDE * 2);                 // [128][260]
    __bf16* Wl   = (__bf16*)(smem + L_ * FES_STRIDE * 2 + L_ * ATT_STRIDE * 4); // [256][40]

    const int j = blockIdx.x, b = blockIdx.y;
    const int tid = threadIdx.x, lane = tid & 31, wv = tid >> 5;
    const unsigned feSB = lds_addr(feS);
    const unsigned wlB  = lds_addr(Wl);

    // async-DMA fe[b, :, j, :] slab: 128 rows x 512B, 16 x b128 per thread
    for (int cidx = tid; cidx < L_ * 32; cidx += 256) {
        int i = cidx >> 5, c = (cidx & 31) * 16;
        async_copy16(feSB + (unsigned)(i * (FES_STRIDE * 2) + c), fe,
                     (unsigned)(((size_t)((b * L_ + i) * L_ + j)) * (H_ * 2) + c));
    }
    async_wait0();
    __syncthreads();

    v8f acc[16] = {};
    const int i0 = wv * 16;   // wave owns 16-row i-strip, full N=256

    for (int kc = 0; kc < H_ / 32; ++kc) {
        const int k0 = kc << 5;
        // async-DMA W_attn[:, k0:k0+32]: 256 rows x 64B, 4 x b128 per thread
        for (int cidx = tid; cidx < H_ * 4; cidx += 256) {
            int n = cidx >> 2, c = (cidx & 3) * 16;
            async_copy16(wlB + (unsigned)(n * (WL_STRIDE * 2) + c), Wattn,
                         (unsigned)((size_t)n * (H_ * 2) + k0 * 2 + c));
        }
        async_wait0();
        __syncthreads();

        v16bf af;
        {
            int m  = i0 + (lane & 15);
            int kb = (lane >> 4) << 3;
            #pragma unroll
            for (int e = 0; e < 16; ++e) {
                int kk = (e < 8) ? (kb + e) : (kb + e + 8);
                af[e] = feS[m * FES_STRIDE + k0 + kk];
            }
        }
        #pragma unroll
        for (int t = 0; t < 16; ++t) {
            v16bf bfrg;
            int n  = t * 16 + (lane & 15);
            int kb = (lane >> 4) << 3;
            #pragma unroll
            for (int e = 0; e < 16; ++e) bfrg[e] = Wl[n * WL_STRIDE + ((e < 8) ? (kb + e) : (kb + e + 8))];
            acc[t] = __builtin_amdgcn_wmma_f32_16x16x32_bf16(
                false, af, false, bfrg, (short)0, acc[t], false, false);
        }
        __syncthreads();
    }

    // spill attn tile (+bias) to LDS
    #pragma unroll
    for (int t = 0; t < 16; ++t) {
        #pragma unroll
        for (int r = 0; r < 8; ++r) {
            int i = i0 + r + ((lane >> 4) << 3);
            int h = t * 16 + (lane & 15);
            attS[i * ATT_STRIDE + h] = acc[t][r] + battn[h];
        }
    }
    __syncthreads();

    // per-channel softmax over i, fused weighted sum and sigmoid
    const int h = tid;
    float mx = -3.4e38f;
    for (int i = 0; i < L_; ++i) mx = fmaxf(mx, attS[i * ATT_STRIDE + h]);
    float s = 0.f, a = 0.f;
    const __bf16* xh = xhid + (size_t)(b * L_) * H_ + h;
    for (int i = 0; i < L_; ++i) {
        float e = __expf(attS[i * ATT_STRIDE + h] - mx);
        s += e;
        a += e * (float)xh[(size_t)i * H_];
    }
    float r = a / s;
    agg[(size_t)(b * L_ + j) * H_ + h] = (__bf16)(1.f / (1.f + __expf(-r)));
}

// ---------------------------------------------------------------------------
extern "C" void kernel_launch(void* const* d_in, const int* in_sizes, int n_in,
                              void* d_out, int out_size, void* d_ws, size_t ws_size,
                              hipStream_t stream)
{
    const float* feat   = (const float*)d_in[0];
    const float* mask   = (const float*)d_in[1];
    const float* W_inp  = (const float*)d_in[2];
    const float* b_inp  = (const float*)d_in[3];
    const float* W_oup  = (const float*)d_in[4];
    const float* b_oup  = (const float*)d_in[5];
    const float* W_fe   = (const float*)d_in[6];
    const float* b_fe   = (const float*)d_in[7];
    const float* W_ue   = (const float*)d_in[8];
    const float* b_ue   = (const float*)d_in[9];
    const float* W_agg  = (const float*)d_in[10];
    const float* b_agg  = (const float*)d_in[11];
    const float* W_uv   = (const float*)d_in[12];
    const float* b_uv   = (const float*)d_in[13];
    const float* W_attn = (const float*)d_in[14];
    const float* b_attn = (const float*)d_in[15];
    float* out = (float*)d_out;

    char* base = (char*)d_ws;
    size_t off = 0;
    auto alloc = [&](size_t bytes) -> char* {
        char* p = base + off; off += (bytes + 255) & ~(size_t)255; return p;
    };
    __bf16* fe_a   = (__bf16*)alloc((size_t)ROWS_E * H_ * 2);
    __bf16* fe_b   = (__bf16*)alloc((size_t)ROWS_E * H_ * 2);
    __bf16* featb  = (__bf16*)alloc((size_t)ROWS_V * NINP_ * 2);
    __bf16* fv     = (__bf16*)alloc((size_t)ROWS_V * H_ * 2);
    __bf16* fv2    = (__bf16*)alloc((size_t)ROWS_V * H_ * 2);
    __bf16* xx     = (__bf16*)alloc((size_t)ROWS_V * H_ * 2);
    __bf16* yy     = (__bf16*)alloc((size_t)ROWS_V * H_ * 2);
    __bf16* px     = (__bf16*)alloc((size_t)ROWS_V * H_ * 2);
    __bf16* py     = (__bf16*)alloc((size_t)ROWS_V * H_ * 2);
    __bf16* xh     = (__bf16*)alloc((size_t)ROWS_V * H_ * 2);
    __bf16* ag     = (__bf16*)alloc((size_t)ROWS_V * H_ * 2);
    __bf16* Wb_inp = (__bf16*)alloc((size_t)H_ * NINP_ * 2);
    __bf16* Wb_fe  = (__bf16*)alloc((size_t)H_ * 2 * H_ * 2);
    __bf16* Wb_ue  = (__bf16*)alloc((size_t)H_ * 2 * H_ * 2);
    __bf16* Wb_agg = (__bf16*)alloc((size_t)H_ * H_ * 2);
    __bf16* Wb_uv  = (__bf16*)alloc((size_t)H_ * 2 * H_ * 2);
    __bf16* Wb_attn= (__bf16*)alloc((size_t)H_ * H_ * 2);
    __bf16* Wb_oup = (__bf16*)alloc((size_t)NINP_ * H_ * 2);
    float*  cv     = (float*)alloc(H_ * 4);

    dim3 blk(256);
    auto cvt = [&](const float* s, __bf16* d, int n) {
        cvt_bf16_kernel<<<dim3((n + 255) / 256), blk, 0, stream>>>(s, d, n);
    };
    auto gemm = [&](const __bf16* A, int lda, const __bf16* W, int ldw,
                    const float* bias, const float* rowscale,
                    void* C, int ldc, int cbf, int M, int N, int K, int mode,
                    const __bf16* pxp = nullptr, const __bf16* pyp = nullptr,
                    const float* cvp = nullptr) {
        dim3 grid(N / 64, M / 64);
        wmma_gemm_kernel<<<grid, blk, 0, stream>>>(A, lda, W, ldw, bias, rowscale,
                                                   C, ldc, cbf, K, mode, pxp, pyp, cvp);
    };

    // one-time f32 -> bf16 residency conversion
    cvt(feat,   featb,  ROWS_V * NINP_);
    cvt(W_inp,  Wb_inp, H_ * NINP_);
    cvt(W_fe,   Wb_fe,  H_ * 2 * H_);
    cvt(W_ue,   Wb_ue,  H_ * 2 * H_);
    cvt(W_agg,  Wb_agg, H_ * H_);
    cvt(W_uv,   Wb_uv,  H_ * 2 * H_);
    cvt(W_attn, Wb_attn,H_ * H_);
    cvt(W_oup,  Wb_oup, NINP_ * H_);

    // fv = feat @ W_inp^T + b_inp
    gemm(featb, NINP_, Wb_inp, NINP_, b_inp, nullptr, fv, H_, 1, ROWS_V, H_, NINP_, GM_PLAIN);

    // fe init
    gemm(fv, H_, Wb_fe,      2 * H_, nullptr, nullptr, xx, H_, 1, ROWS_V, H_, H_, GM_PLAIN);
    gemm(fv, H_, Wb_fe + H_, 2 * H_, nullptr, nullptr, yy, H_, 1, ROWS_V, H_, H_, GM_PLAIN);
    outer_sum_kernel<<<dim3((ROWS_E * H_) / 256), blk, 0, stream>>>(xx, yy, b_fe, fe_a);

    __bf16* fe_src = fe_a;  __bf16* fe_dst = fe_b;
    __bf16* fv_cur = fv;    __bf16* fv_nxt = fv2;

    for (int step = 0; step < 3; ++step) {
        // edge update, factored: fe' = fe@Wue1^T + px_i + py_j + cvec
        gemm(fv_cur, H_, Wb_fe,      2 * H_, nullptr, nullptr, xx, H_, 1, ROWS_V, H_, H_, GM_PLAIN);
        gemm(fv_cur, H_, Wb_fe + H_, 2 * H_, nullptr, nullptr, yy, H_, 1, ROWS_V, H_, H_, GM_PLAIN);
        gemm(xx, H_, Wb_ue, 2 * H_, nullptr, nullptr, px, H_, 1, ROWS_V, H_, H_, GM_PLAIN);
        gemm(yy, H_, Wb_ue, 2 * H_, nullptr, nullptr, py, H_, 1, ROWS_V, H_, H_, GM_PLAIN);
        cvec_kernel<<<dim3(1), blk, 0, stream>>>(b_fe, W_ue, b_ue, cv);
        gemm(fe_src, H_, Wb_ue + H_, 2 * H_, nullptr, nullptr, fe_dst, H_, 1,
             ROWS_E, H_, H_, GM_EDGE, px, py, cv);
        { __bf16* t = fe_src; fe_src = fe_dst; fe_dst = t; }

        // vertex update
        gemm(fv_cur, H_, Wb_agg, H_, b_agg, mask, xh, H_, 1, ROWS_V, H_, H_, GM_PLAIN);
        attn_softmax_agg_kernel<<<dim3(L_, B_), blk, ATTN_LDS_BYTES, stream>>>(
            fe_src, Wb_attn, b_attn, xh, ag);
        gemm(xh, H_, Wb_uv,      2 * H_, b_uv,   nullptr, fv_nxt, H_, 1, ROWS_V, H_, H_, GM_PLAIN);
        gemm(ag, H_, Wb_uv + H_, 2 * H_, nullptr, nullptr, fv_nxt, H_, 1, ROWS_V, H_, H_, GM_ACC);
        { __bf16* t = fv_cur; fv_cur = fv_nxt; fv_nxt = t; }
    }

    // out = fv @ W_oup^T + b_oup (f32 output)
    gemm(fv_cur, H_, Wb_oup, H_, b_oup, nullptr, out, NINP_, 0, ROWS_V, NINP_, H_, GM_PLAIN);

    (void)in_sizes; (void)n_in; (void)out_size; (void)ws_size;
}
